// MultiheadAttention_36996848287766
// MI455X (gfx1250) — matrix operand order repaired and checked by reading, not compiled, not run on hardware
//
#include <hip/hip_runtime.h>
#include <hip/hip_bf16.h>

typedef __attribute__((ext_vector_type(16))) __bf16 v16bf;
typedef __attribute__((ext_vector_type(8)))  float  v8f;
typedef __attribute__((ext_vector_type(4)))  unsigned u32x4;
typedef __attribute__((ext_vector_type(8)))  unsigned u32x8;

#define SEQ   2048
#define BSZ   4
#define EMB   512
#define NH    8
#define HD    64
#define ROWS  (SEQ*BSZ)      // 8192
#define BH    (BSZ*NH)       // 32
#define WPB   8              // waves per 256-thread block
#define KBLK  32             // keys per iteration
#define NITER (SEQ/KBLK)     // 64

static __device__ __forceinline__ v8f wmma_bf16(v16bf a, v16bf b, v8f c) {
    return __builtin_amdgcn_wmma_f32_16x16x32_bf16(false, a, false, b, (short)0, c, false, false);
}

// A-matrix fragment (16x32, bf16), contiguous k order (matches B loads):
// lane half 0: elems 0..15 -> K = k0+j
// lane half 1: elems 0..15 -> K = k0+16+j
static __device__ __forceinline__ v16bf load_a_f32(const float* row_ptr, int k0, int half) {
    v16bf a;
    const float* p0 = row_ptr + k0 + half*16;
    const float* p1 = row_ptr + k0 + 8 + half*16;
    #pragma unroll
    for (int j = 0; j < 8; ++j) { a[j] = (__bf16)p0[j]; a[8+j] = (__bf16)p1[j]; }
    return a;
}
static __device__ __forceinline__ v16bf load_a_bf16(const __bf16* row_ptr, int k0, int half) {
    v16bf a;
    const __bf16* p0 = row_ptr + k0 + half*16;
    const __bf16* p1 = row_ptr + k0 + 8 + half*16;
    #pragma unroll
    for (int j = 0; j < 8; ++j) { a[j] = p0[j]; a[8+j] = p1[j]; }
    return a;
}

// ---------------------------------------------------------------------------
// TDM: DMA a 2D bf16 tile (tile_d0 x tile_d1 rows) from global into LDS.
// Descriptor layout per CDNA5 ISA sec. 8 (D# groups 0/1; 2D so groups 2/3 NULL).
// Issued uniformly (values are workgroup-uniform); tracked by TENSORcnt.
// ---------------------------------------------------------------------------
static __device__ __forceinline__ void tdm_load_2d(const void* gptr, unsigned lds_addr,
                                                   unsigned dim0, unsigned dim1,
                                                   unsigned tile0, unsigned tile1,
                                                   unsigned stride0)
{
    const unsigned long long ga = (unsigned long long)gptr;
    u32x4 g0;
    g0[0] = 1u;                                            // count=1, user mode
    g0[1] = lds_addr;                                      // LDS byte address
    g0[2] = (unsigned)ga;                                  // global addr [31:0]
    g0[3] = (unsigned)((ga >> 32) & 0x01FFFFFFu) | (2u << 30); // addr[56:32] | type=2
    u32x8 g1;
    g1[0] = (1u << 16);                                    // data_size=1 (2 bytes)
    g1[1] = (dim0 & 0xFFFFu) << 16;                        // tensor_dim0[15:0]
    g1[2] = (dim0 >> 16) | ((dim1 & 0xFFFFu) << 16);       // dim0 hi | tensor_dim1 lo
    g1[3] = (dim1 >> 16) | (tile0 << 16);                  // dim1 hi | tile_dim0
    g1[4] = tile1;                                         // tile_dim1 | tile_dim2=0
    g1[5] = stride0;                                       // tensor_dim0_stride lo
    g1[6] = 0u;                                            // stride0 hi | stride1 lo
    g1[7] = 0u;                                            // stride1 hi
    asm volatile("tensor_load_to_lds %0, %1" :: "s"(g0), "s"(g1) : "memory");
}

// ---------------------------------------------------------------------------
// Weight f32 -> bf16 conversion
// ---------------------------------------------------------------------------
__global__ void __launch_bounds__(256)
convert_kernel(const float* __restrict__ Win, const float* __restrict__ Wout,
               __bf16* __restrict__ WinBf, __bf16* __restrict__ WoutBf)
{
    const int n1 = 3*EMB*EMB;
    const int nt = n1 + EMB*EMB;
    for (int i = blockIdx.x*blockDim.x + threadIdx.x; i < nt; i += gridDim.x*blockDim.x) {
        if (i < n1) WinBf[i] = (__bf16)Win[i];
        else        WoutBf[i-n1] = (__bf16)Wout[i-n1];
    }
}

// ---------------------------------------------------------------------------
// Fused in-projection: [8192 x 512] x [1536 x 512]^T + bias.
//   Q  -> [B*H, S, 64] bf16, pre-scaled by 1/sqrt(64)
//   K  -> [B*H, S, 64] bf16
//   Vt -> [B*H, 64, S] bf16 (transposed: key contiguous for PV B-frags)
// ---------------------------------------------------------------------------
__global__ void __launch_bounds__(256)
inproj_kernel(const float* __restrict__ q_in, const float* __restrict__ k_in,
              const float* __restrict__ v_in, const __bf16* __restrict__ Wbf,
              const float* __restrict__ bias,
              __bf16* __restrict__ Q, __bf16* __restrict__ Kd, __bf16* __restrict__ Vt)
{
    const int lane = threadIdx.x & 31;
    const int half = lane >> 4;
    const int lm   = lane & 15;
    const int wave = (blockIdx.x * blockDim.x + threadIdx.x) >> 5;
    const int CT = (3*EMB)/16;                // 96 col tiles
    const int rt = wave / CT;
    const int ct = wave % CT;
    if (rt >= ROWS/16) return;
    const int r0 = rt*16, o0 = ct*16;
    const int which = o0 / EMB;               // 0=q 1=k 2=v
    const float* X = (which==0) ? q_in : ((which==1) ? k_in : v_in);

    const float*  arow = X + (size_t)(r0 + lm)*EMB;
    const __bf16* brow = Wbf + (size_t)(o0 + lm)*EMB + half*16;

    v8f acc = {0.f,0.f,0.f,0.f,0.f,0.f,0.f,0.f};
    #pragma unroll 4
    for (int k0 = 0; k0 < EMB; k0 += 32) {
        v16bf a = load_a_f32(arow, k0, half);
        v16bf b = *(const v16bf*)(brow + k0);
        acc = wmma_bf16(a, b, acc);
    }

    const int   o    = o0 + lm;
    const float bval = bias[o];
    const int   oc   = o % EMB;
    const int   h    = oc >> 6;
    const int   d    = oc & 63;
    const float scale = (which==0) ? 0.125f : 1.0f;   // 1/sqrt(head_dim)
    #pragma unroll
    for (int r = 0; r < 8; ++r) {
        const int gr = r0 + r + half*8;       // flat row = s*BSZ + b
        const int s  = gr >> 2;
        const int bb = gr & 3;
        const int bh = bb*NH + h;
        const float vv = (acc[r] + bval) * scale;
        if (which == 2) {
            Vt[((size_t)bh*HD + d)*SEQ + s] = (__bf16)vv;           // [bh][d][key]
        } else {
            __bf16* dst = (which==0) ? Q : Kd;
            dst[((size_t)bh*SEQ + s)*HD + d] = (__bf16)vv;          // [bh][s][d]
        }
    }
}

// ---------------------------------------------------------------------------
// Flash attention, block-cooperative: 8 waves of a block share one head and
// 8 consecutive 16-row query tiles. The 32-key K/V tiles are DMA'd into LDS
// by the Tensor Data Mover (double-buffered, TENSORcnt-tracked); all WMMA
// B-fragments then come from 32B LDS vector loads. Online softmax in f32.
// ---------------------------------------------------------------------------
__global__ void __launch_bounds__(256)
attn_kernel(const __bf16* __restrict__ Q, const __bf16* __restrict__ K,
            const __bf16* __restrict__ Vt, float* __restrict__ a_test,
            __bf16* __restrict__ a_bf)
{
    __shared__ __bf16 Klds[2][KBLK*HD];       // 2 x 4KB: [key][d]
    __shared__ __bf16 Vlds[2][HD*KBLK];       // 2 x 4KB: [d][key]
    __shared__ __bf16 lds_p[WPB][16*32];      // per-wave P staging (C->A remap)

    const int lane  = threadIdx.x & 31;
    const int half  = lane >> 4;
    const int lm    = lane & 15;
    const int wslot = threadIdx.x >> 5;

    const int bh = blockIdx.x >> 4;           // 32 heads
    const int qb = blockIdx.x & 15;           // 16 query blocks of 128 rows
    const int qt = qb*WPB + wslot;            // this wave's 16-row query tile

    const __bf16* Qh = Q  + (size_t)bh*SEQ*HD;
    const __bf16* Kh = K  + (size_t)bh*SEQ*HD;
    const __bf16* Vh = Vt + (size_t)bh*HD*SEQ;   // [d][key]

    const __bf16* qrow = Qh + (size_t)(qt*16 + lm)*HD;
    const v16bf qa0 = load_a_bf16(qrow, 0, half);
    const v16bf qa1 = load_a_bf16(qrow, 32, half);

    v8f oacc[4];
    float m[8], l[8];
    #pragma unroll
    for (int r = 0; r < 8; ++r) { m[r] = -1e30f; l[r] = 0.f; }
    #pragma unroll
    for (int t = 0; t < 4; ++t)
        #pragma unroll
        for (int r = 0; r < 8; ++r) oacc[t][r] = 0.f;

    // ---- TDM prologue: stage first K/V tiles into buffer 0 -----------------
    if (wslot == 0) {
        tdm_load_2d(Kh, (unsigned)(size_t)&Klds[0][0],
                    HD, KBLK, HD, KBLK, HD);            // 32 rows x 64, stride 64
        tdm_load_2d(Vh, (unsigned)(size_t)&Vlds[0][0],
                    KBLK, HD, KBLK, HD, SEQ);           // 64 rows x 32, stride 2048
    }

    for (int i = 0; i < NITER; ++i) {
        const int cur = i & 1;
        if (wslot == 0) {
            if (i + 1 < NITER) {
                const int nxt = cur ^ 1;
                const int kbn = (i + 1) * KBLK;
                tdm_load_2d(Kh + (size_t)kbn*HD, (unsigned)(size_t)&Klds[nxt][0],
                            HD, KBLK, HD, KBLK, HD);
                tdm_load_2d(Vh + kbn, (unsigned)(size_t)&Vlds[nxt][0],
                            KBLK, HD, KBLK, HD, SEQ);
                __builtin_amdgcn_s_wait_tensorcnt(2);   // current buffer done
            } else {
                __builtin_amdgcn_s_wait_tensorcnt(0);
            }
        }
        __syncthreads();                                 // cur buffer valid for all
        const __bf16* Kc = &Klds[cur][0];
        const __bf16* Vc = &Vlds[cur][0];

        // ---- scores S = (Q/sqrt(d)) K^T for 32 keys: two 16x16 C tiles ----
        v8f s0 = {0.f,0.f,0.f,0.f,0.f,0.f,0.f,0.f};
        v8f s1 = {0.f,0.f,0.f,0.f,0.f,0.f,0.f,0.f};
        {
            const __bf16* kr0 = Kc + lm*HD + half*16;
            v16bf b00 = *(const v16bf*)(kr0);
            v16bf b01 = *(const v16bf*)(kr0 + 32);
            s0 = wmma_bf16(qa0, b00, s0);
            s0 = wmma_bf16(qa1, b01, s0);
            const __bf16* kr1 = Kc + (16 + lm)*HD + half*16;
            v16bf b10 = *(const v16bf*)(kr1);
            v16bf b11 = *(const v16bf*)(kr1 + 32);
            s1 = wmma_bf16(qa0, b10, s1);
            s1 = wmma_bf16(qa1, b11, s1);
        }

        // ---- online softmax update (row reductions across 16-lane halves) --
        float corr[8];
        #pragma unroll
        for (int r = 0; r < 8; ++r) {
            float mx = fmaxf(s0[r], s1[r]);
            #pragma unroll
            for (int off = 8; off; off >>= 1) mx = fmaxf(mx, __shfl_xor(mx, off, 32));
            const float mn = fmaxf(m[r], mx);
            corr[r] = __expf(m[r] - mn);
            m[r] = mn;
            const float p0 = __expf(s0[r] - mn);
            const float p1 = __expf(s1[r] - mn);
            s0[r] = p0; s1[r] = p1;
            float sum = p0 + p1;
            #pragma unroll
            for (int off = 8; off; off >>= 1) sum += __shfl_xor(sum, off, 32);
            l[r] = l[r]*corr[r] + sum;
        }
        #pragma unroll
        for (int t = 0; t < 4; ++t)
            #pragma unroll
            for (int r = 0; r < 8; ++r) oacc[t][r] *= corr[r];

        // ---- remap P from C-tile layout to A-fragment layout via LDS -------
        __bf16* lp = lds_p[wslot];
        #pragma unroll
        for (int r = 0; r < 8; ++r) {
            const int M = r + half*8;
            lp[M*32 + lm]      = (__bf16)s0[r];
            lp[M*32 + 16 + lm] = (__bf16)s1[r];
        }
        asm volatile("s_wait_dscnt 0" ::: "memory");
        const v16bf pa = load_a_bf16(lp + lm*32, 0, half);

        // ---- O += P @ V: B-frags from LDS V tile ([d][key], key contig) ----
        #pragma unroll
        for (int t = 0; t < 4; ++t) {
            const __bf16* vp = Vc + (t*16 + lm)*KBLK + half*16;
            v16bf vb = *(const v16bf*)vp;
            oacc[t] = wmma_bf16(pa, vb, oacc[t]);
        }
        __syncthreads();   // all waves done with cur before TDM overwrites it
    }

    // ---- finalize: O /= l ; scatter to a_test [s*B+b][h*64+d] --------------
    const int b = bh >> 3, h = bh & 7;
    #pragma unroll
    for (int r = 0; r < 8; ++r) {
        const float inv  = 1.0f / l[r];
        const int   srow = qt*16 + r + half*8;
        const size_t base = (size_t)(srow*BSZ + b)*EMB + h*HD;
        #pragma unroll
        for (int t = 0; t < 4; ++t) {
            const float vv = oacc[t][r] * inv;
            const size_t idx = base + t*16 + lm;
            __builtin_nontemporal_store(vv, &a_test[idx]);  // write-only stream
            a_bf[idx] = (__bf16)vv;
        }
    }
}

// ---------------------------------------------------------------------------
// Out-projection: [8192 x 512] x [512 x 512]^T + bias -> attn_output (f32)
// ---------------------------------------------------------------------------
__global__ void __launch_bounds__(256)
outproj_kernel(const __bf16* __restrict__ A, const __bf16* __restrict__ Wbf,
               const float* __restrict__ bias, float* __restrict__ out)
{
    const int lane = threadIdx.x & 31;
    const int half = lane >> 4;
    const int lm   = lane & 15;
    const int wave = (blockIdx.x * blockDim.x + threadIdx.x) >> 5;
    const int CT = EMB/16;                    // 32
    const int rt = wave / CT;
    const int ct = wave % CT;
    if (rt >= ROWS/16) return;
    const int r0 = rt*16, o0 = ct*16;

    const __bf16* arow = A   + (size_t)(r0 + lm)*EMB;
    const __bf16* brow = Wbf + (size_t)(o0 + lm)*EMB + half*16;

    v8f acc = {0.f,0.f,0.f,0.f,0.f,0.f,0.f,0.f};
    #pragma unroll 4
    for (int k0 = 0; k0 < EMB; k0 += 32) {
        v16bf a = load_a_bf16(arow, k0, half);
        v16bf b = *(const v16bf*)(brow + k0);
        acc = wmma_bf16(a, b, acc);
    }

    const float bval = bias[o0 + lm];
    #pragma unroll
    for (int r = 0; r < 8; ++r) {
        const float vv = acc[r] + bval;
        __builtin_nontemporal_store(vv, &out[(size_t)(r0 + r + half*8)*EMB + o0 + lm]);
    }
}

// ---------------------------------------------------------------------------
extern "C" void kernel_launch(void* const* d_in, const int* in_sizes, int n_in,
                              void* d_out, int out_size, void* d_ws, size_t ws_size,
                              hipStream_t stream) {
    const float* query  = (const float*)d_in[0];
    const float* key_   = (const float*)d_in[1];
    const float* value  = (const float*)d_in[2];
    const float* Wi     = (const float*)d_in[3];   // [3E, E]
    const float* bi     = (const float*)d_in[4];   // [3E]
    const float* Wo     = (const float*)d_in[5];   // [E, E]
    const float* bo     = (const float*)d_in[6];   // [E]

    float* out    = (float*)d_out;                 // attn_output [S,B,E]
    float* a_test = out + (size_t)ROWS*EMB;        // a_test [S*B, E]

    char* w = (char*)d_ws;
    __bf16* WinBf  = (__bf16*)w;  w += (size_t)3*EMB*EMB*sizeof(__bf16);
    __bf16* WoutBf = (__bf16*)w;  w += (size_t)EMB*EMB*sizeof(__bf16);
    __bf16* Qbf    = (__bf16*)w;  w += (size_t)BH*SEQ*HD*sizeof(__bf16);
    __bf16* Kbf    = (__bf16*)w;  w += (size_t)BH*SEQ*HD*sizeof(__bf16);
    __bf16* Vtbf   = (__bf16*)w;  w += (size_t)BH*HD*SEQ*sizeof(__bf16);
    __bf16* Abf    = (__bf16*)w;  w += (size_t)ROWS*EMB*sizeof(__bf16);

    convert_kernel<<<1024, 256, 0, stream>>>(Wi, Wo, WinBf, WoutBf);

    // 512 row-tiles * 96 col-tiles = 49152 waves / 8 per block
    inproj_kernel<<<(ROWS/16)*((3*EMB)/16)/WPB, 256, 0, stream>>>(
        query, key_, value, WinBf, bi, Qbf, Kbf, Vtbf);

    // 32 heads * 16 query-blocks (8 waves x 16 rows each) = 512 blocks
    attn_kernel<<<BH*(SEQ/(16*WPB)), 256, 0, stream>>>(Qbf, Kbf, Vtbf, a_test, Abf);

    // 512 row-tiles * 32 col-tiles = 16384 waves / 8 per block
    outproj_kernel<<<(ROWS/16)*(EMB/16)/WPB, 256, 0, stream>>>(Abf, WoutBf, bo, out);
}
